// MyTransformerXL_46866683134054
// MI455X (gfx1250) — matrix operand order repaired and checked by reading, not compiled, not run on hardware
//
#include <hip/hip_runtime.h>
#include <math.h>

#define SEG    1024
#define BSZ    4
#define DM     512
#define NH     8
#define DH     64
#define KLEN   2048
#define MEMN   1024
#define DI     2048
#define NLEADS 12
#define NLAYER 4

typedef _Float16 half_t;
typedef __attribute__((ext_vector_type(16))) _Float16 v16h;
typedef __attribute__((ext_vector_type(8)))  float    v8f;

// ---------------------------------------------------------------------------
// WMMA helpers (CDNA5: v_wmma_f32_16x16x32_f16, wave32, f32 accumulate)
// ---------------------------------------------------------------------------
static __device__ inline v8f wmma16(v16h a, v16h b, v8f c) {
  return __builtin_amdgcn_wmma_f32_16x16x32_f16(false, a, false, b, (short)0, c,
                                                false, false);
}

// A fragment (16x32, MxK) from row-major f32. Lane L: row = L&15,
// K = 16*(L>>4) + e (contiguous, matches B fragments).
static __device__ inline v16h a_frag_f32(const float* A, int lda, int lane) {
  int m = lane & 15;
  int kb = (lane >> 4) * 16;
  const float* r = A + (long)m * lda;
  v16h a;
#pragma unroll
  for (int e = 0; e < 8; ++e) {
    a[e]     = (_Float16)r[kb + e];
    a[e + 8] = (_Float16)r[kb + 8 + e];
  }
  return a;
}

static __device__ inline v16h a_frag_f16(const half_t* A, int lda, int lane) {
  int m = lane & 15;
  int kb = (lane >> 4) * 16;
  const half_t* r = A + (long)m * lda;
  v16h a;
#pragma unroll
  for (int e = 0; e < 8; ++e) {
    a[e]     = r[kb + e];
    a[e + 8] = r[kb + 8 + e];
  }
  return a;
}

// B fragment (32x16, KxN) from row-major f32 B[k][n]. Lane L: col = L&15,
// K = 16*(L>>4) + e. Contiguous, same k order as A fragments.
static __device__ inline v16h b_frag_f32(const float* B, int ldb, int lane) {
  int n = lane & 15;
  int kb = (lane >> 4) * 16;
  v16h b;
#pragma unroll
  for (int e = 0; e < 16; ++e) b[e] = (_Float16)B[(long)(kb + e) * ldb + n];
  return b;
}

static __device__ inline v16h b_frag_f16(const half_t* B, int ldb, int lane) {
  int n = lane & 15;
  int kb = (lane >> 4) * 16;
  v16h b;
#pragma unroll
  for (int e = 0; e < 16; ++e) b[e] = B[(long)(kb + e) * ldb + n];
  return b;
}

// B fragment where source rows are the *N* dimension (K contiguous):
// B[k][n] = src[n*rstride + k]  (i.e. an implicit transpose fetch: S = Q*K^T).
static __device__ inline v16h bt_frag_f16(const half_t* src, int rstride, int lane) {
  int n = lane & 15;
  int kb = (lane >> 4) * 16;
  const half_t* p = src + (long)n * rstride + kb;
  v16h b;
#pragma unroll
  for (int e = 0; e < 16; ++e) b[e] = p[e];
  return b;
}

// Same but with row-clamp for the rel-shift rk window (p>=KLEN is masked later)
static __device__ inline v16h bt_frag_rk(const half_t* Rk, int prow0, int coloff,
                                         int lane) {
  int n = lane & 15;
  int kb = (lane >> 4) * 16;
  int p = prow0 + n;
  if (p >= KLEN) p = 0;  // value unused: corresponds exactly to masked j > i+MEM
  const half_t* q = Rk + (long)p * DM + coloff + kb;
  v16h b;
#pragma unroll
  for (int e = 0; e < 16; ++e) b[e] = q[e];
  return b;
}

// ---------------------------------------------------------------------------
// x = data @ Wemb + bemb ; also store hids[0]
// ---------------------------------------------------------------------------
__global__ void txl_embed_kernel(const float* __restrict__ data,
                                 const float* __restrict__ Wemb,
                                 const float* __restrict__ bemb,
                                 float* __restrict__ x, float* __restrict__ hid0) {
  int row = blockIdx.x;
  const float* dr = data + (long)row * NLEADS;
  for (int c = threadIdx.x; c < DM; c += 256) {
    float acc = bemb[c];
#pragma unroll
    for (int l = 0; l < NLEADS; ++l) acc += dr[l] * Wemb[l * DM + c];
    x[(long)row * DM + c] = acc;
    hid0[(long)row * DM + c] = acc;
  }
}

// sinusoidal relative positions pos[p][c], pos_seq descending KLEN-1..0
__global__ void txl_pos_kernel(float* __restrict__ pos) {
  int idx = blockIdx.x * 256 + threadIdx.x;  // KLEN*DM total
  int p = idx >> 9;
  int c = idx & 511;
  int ci = c & 255;
  // inv_freq = 10000^(-2c/512) = exp(-c * 2*ln(10000)/512)
  float freq = __expf(-0.03597789f * (float)ci);
  float ang = (float)(KLEN - 1 - p) * freq;
  pos[idx] = (c < 256) ? __sinf(ang) : __cosf(ang);
}

// ---------------------------------------------------------------------------
// QKV projection: [mems_i ; x] (8192 x 512) @ W (512 x 1536)
// epilogue: Qa=(q+r_w_bias), Qb=(q+r_r_bias) for last SEG rows; K, V as f16.
// ---------------------------------------------------------------------------
__global__ void txl_qkv_gemm_kernel(const float* __restrict__ mems_i,
                                    const float* __restrict__ x,
                                    const float* __restrict__ W,
                                    const float* __restrict__ rwb,
                                    const float* __restrict__ rrb,
                                    half_t* __restrict__ Qa, half_t* __restrict__ Qb,
                                    half_t* __restrict__ Kf, half_t* __restrict__ Vf) {
  int lane = threadIdx.x;
  int n0 = blockIdx.x * 64;
  int m0 = blockIdx.y * 16;
  const int Msplit = MEMN * BSZ;  // 4096
  v8f acc[4] = {};

  int m = m0 + (lane & 15);
  const float* arow =
      (m < Msplit) ? (mems_i + (long)m * DM) : (x + (long)(m - Msplit) * DM);
  int kb = (lane >> 4) * 16;

  for (int k0 = 0; k0 < DM; k0 += 32) {
    if (k0 + 32 < DM) __builtin_prefetch(arow + k0 + 32, 0, 1);
    v16h a;
#pragma unroll
    for (int e = 0; e < 8; ++e) {
      a[e]     = (_Float16)arow[k0 + kb + e];
      a[e + 8] = (_Float16)arow[k0 + kb + 8 + e];
    }
#pragma unroll
    for (int t = 0; t < 4; ++t) {
      v16h b = b_frag_f32(W + (long)k0 * 1536 + n0 + t * 16, 1536, lane);
      acc[t] = wmma16(a, b, acc[t]);
    }
  }

  int nn = lane & 15, rh = (lane >> 4) * 8;
#pragma unroll
  for (int t = 0; t < 4; ++t) {
    int col = n0 + t * 16 + nn;
    int qt = col / DM;        // 0=q, 1=k, 2=v
    int cc = col - qt * DM;   // head*64 + d
#pragma unroll
    for (int r = 0; r < 8; ++r) {
      int row = m0 + rh + r;
      float v = acc[t][r];
      if (qt == 0) {
        if (row >= Msplit) {
          long qr = (long)(row - Msplit) * DM + cc;
          Qa[qr] = (_Float16)(v + rwb[cc]);
          Qb[qr] = (_Float16)(v + rrb[cc]);
        }
      } else if (qt == 1) {
        Kf[(long)row * DM + cc] = (_Float16)v;
      } else {
        Vf[(long)row * DM + cc] = (_Float16)v;
      }
    }
  }
}

// rk = pos (2048x512) @ r_w_i (512x512), f16 out
__global__ void txl_rk_gemm_kernel(const float* __restrict__ pos,
                                   const float* __restrict__ W,
                                   half_t* __restrict__ Rk) {
  int lane = threadIdx.x;
  int n0 = blockIdx.x * 64;
  int m0 = blockIdx.y * 16;
  v8f acc[4] = {};
  for (int k0 = 0; k0 < DM; k0 += 32) {
    v16h a = a_frag_f32(pos + (long)m0 * DM + k0, DM, lane);
#pragma unroll
    for (int t = 0; t < 4; ++t) {
      v16h b = b_frag_f32(W + (long)k0 * DM + n0 + t * 16, DM, lane);
      acc[t] = wmma16(a, b, acc[t]);
    }
  }
  int nn = lane & 15, rh = (lane >> 4) * 8;
#pragma unroll
  for (int t = 0; t < 4; ++t)
#pragma unroll
    for (int r = 0; r < 8; ++r)
      Rk[(long)(m0 + rh + r) * DM + n0 + t * 16 + nn] = (_Float16)acc[t][r];
}

// ---------------------------------------------------------------------------
// Generic C = act(A@B + bias), f32 in/out, one wave per 16x64 tile
// ---------------------------------------------------------------------------
__global__ void txl_lin_gemm_kernel(const float* __restrict__ A,
                                    const float* __restrict__ B,
                                    const float* __restrict__ bias,
                                    float* __restrict__ C, int M, int N, int K,
                                    int relu) {
  int lane = threadIdx.x;
  int n0 = blockIdx.x * 64;
  int m0 = blockIdx.y * 16;
  v8f acc[4] = {};
  for (int k0 = 0; k0 < K; k0 += 32) {
    if (k0 + 32 < K) __builtin_prefetch(A + (long)m0 * K + k0 + 32, 0, 1);
    v16h a = a_frag_f32(A + (long)m0 * K + k0, K, lane);
#pragma unroll
    for (int t = 0; t < 4; ++t) {
      v16h b = b_frag_f32(B + (long)k0 * N + n0 + t * 16, N, lane);
      acc[t] = wmma16(a, b, acc[t]);
    }
  }
  int nn = lane & 15, rh = (lane >> 4) * 8;
#pragma unroll
  for (int t = 0; t < 4; ++t) {
    int col = n0 + t * 16 + nn;
    float bb = bias ? bias[col] : 0.f;
#pragma unroll
    for (int r = 0; r < 8; ++r) {
      float v = acc[t][r] + bb;
      if (relu) v = fmaxf(v, 0.f);
      C[(long)(m0 + rh + r) * N + col] = v;
    }
  }
}

// ---------------------------------------------------------------------------
// Flash attention with Transformer-XL relative shift.
// One wave per (b, head, 16-query tile); online softmax over k_len=2048.
// BD[i,j] = (q_i + r_r_bias) . rk[j + SEG-1-i]; p >= KLEN <=> masked.
// ---------------------------------------------------------------------------
__global__ void txl_attn_kernel(const half_t* __restrict__ Qa,
                                const half_t* __restrict__ Qb,
                                const half_t* __restrict__ Kf,
                                const half_t* __restrict__ Vf,
                                const half_t* __restrict__ Rk,
                                float* __restrict__ vec) {
  __shared__ float s_lds[16][32];    // AC tile
  __shared__ float raw_lds[16][48];  // Qb.Rk^T window (p0..p0+47)
  __shared__ float p_lds[16][32];    // probabilities
  __shared__ float m_s[16], l_s[16], a_s[16];

  int lane = threadIdx.x;
  int i0 = blockIdx.x * 16;
  int n = blockIdx.y & 7;
  int b = blockIdx.y >> 3;
  int coff = n * DH;
  const int qstride = BSZ * DM;  // stride between consecutive token rows

  // Q fragments for K-chunks 0..31 and 32..63 (D_HEAD = 64)
  const half_t* qabase = Qa + ((long)i0 * BSZ + b) * DM + coff;
  const half_t* qbbase = Qb + ((long)i0 * BSZ + b) * DM + coff;
  v16h qa0 = a_frag_f16(qabase, qstride, lane);
  v16h qa1 = a_frag_f16(qabase + 32, qstride, lane);
  v16h qb0 = a_frag_f16(qbbase, qstride, lane);
  v16h qb1 = a_frag_f16(qbbase + 32, qstride, lane);

  if (lane < 16) { m_s[lane] = -INFINITY; l_s[lane] = 0.f; }
  __syncthreads();

  v8f out[4] = {};
  int nchunks = (i0 + 15 + MEMN) / 32 + 1;  // covers all j <= i+MEM, <= 64

  for (int c = 0; c < nchunks; ++c) {
    int j0 = c * 32;
    // ---- AC = (q + r_w_bias) . K^T, 16x32 as two 16x16 tiles
    v8f s0 = {}, s1 = {};
    {
      const half_t* k0p = Kf + ((long)(j0)*BSZ + b) * DM + coff;
      const half_t* k1p = Kf + ((long)(j0 + 16) * BSZ + b) * DM + coff;
      s0 = wmma16(qa0, bt_frag_f16(k0p, qstride, lane), s0);
      s0 = wmma16(qa1, bt_frag_f16(k0p + 32, qstride, lane), s0);
      s1 = wmma16(qa0, bt_frag_f16(k1p, qstride, lane), s1);
      s1 = wmma16(qa1, bt_frag_f16(k1p + 32, qstride, lane), s1);
    }
    // ---- raw BD window: p in [p0, p0+48), p0 = j0 + SEG-16-i0 >= 0
    int p0 = j0 + SEG - 16 - i0;
    v8f rw[3];
#pragma unroll
    for (int t = 0; t < 3; ++t) {
      v8f r = {};
      r = wmma16(qb0, bt_frag_rk(Rk, p0 + t * 16, coff, lane), r);
      r = wmma16(qb1, bt_frag_rk(Rk, p0 + t * 16, coff + 32, lane), r);
      rw[t] = r;
    }
    // ---- stage tiles (C layout: lane holds col L&15, rows rh..rh+7)
    {
      int nn = lane & 15, rh = (lane >> 4) * 8;
#pragma unroll
      for (int r = 0; r < 8; ++r) {
        s_lds[rh + r][nn] = s0[r];
        s_lds[rh + r][16 + nn] = s1[r];
        raw_lds[rh + r][nn] = rw[0][r];
        raw_lds[rh + r][16 + nn] = rw[1][r];
        raw_lds[rh + r][32 + nn] = rw[2][r];
      }
    }
    __syncthreads();

    // ---- per-row online softmax (16 rows on lanes 0..15)
    if (lane < 16) {
      int i = i0 + lane;
      float mold = m_s[lane];
      float mx = mold;
      float sv[32];
#pragma unroll
      for (int jj = 0; jj < 32; ++jj) {
        int j = j0 + jj;
        float v = -INFINITY;
        if (j <= i + MEMN)  // causal mask with memory offset
          v = (s_lds[lane][jj] + raw_lds[lane][jj + 15 - lane]) * 0.125f;
        sv[jj] = v;
        mx = fmaxf(mx, v);
      }
      float alpha = __expf(mold - mx);  // 0 on first chunk (mold=-inf)
      float lsum = 0.f;
#pragma unroll
      for (int jj = 0; jj < 32; ++jj) {
        float p = __expf(sv[jj] - mx);
        p_lds[lane][jj] = p;
        lsum += p;
      }
      l_s[lane] = l_s[lane] * alpha + lsum;
      m_s[lane] = mx;
      a_s[lane] = alpha;
    }
    __syncthreads();

    // ---- rescale accumulators, P@V (K=32 -> single chained WMMA per N tile)
    {
      int rh = (lane >> 4) * 8;
#pragma unroll
      for (int t = 0; t < 4; ++t)
#pragma unroll
        for (int r = 0; r < 8; ++r) out[t][r] *= a_s[rh + r];

      v16h pf;
      int m = lane & 15, kb2 = (lane >> 4) * 16;
#pragma unroll
      for (int e = 0; e < 8; ++e) {
        pf[e]     = (_Float16)p_lds[m][kb2 + e];
        pf[e + 8] = (_Float16)p_lds[m][kb2 + 8 + e];
      }
      const half_t* vbase = Vf + ((long)j0 * BSZ + b) * DM + coff;
#pragma unroll
      for (int t = 0; t < 4; ++t)
        out[t] = wmma16(pf, b_frag_f16(vbase + t * 16, qstride, lane), out[t]);
    }
    __syncthreads();
  }

  // ---- normalize and write vec (seg, b, nh*dh)
  int nn = lane & 15, rh = (lane >> 4) * 8;
#pragma unroll
  for (int t = 0; t < 4; ++t)
#pragma unroll
    for (int r = 0; r < 8; ++r) {
      int i = i0 + rh + r;
      vec[((long)i * BSZ + b) * DM + coff + t * 16 + nn] =
          out[t][r] / l_s[rh + r];
    }
}

// ---------------------------------------------------------------------------
// out = LayerNorm(xin + dlt) * g + beta ; optional mirror copy (hids slot)
// ---------------------------------------------------------------------------
__global__ void txl_add_ln_kernel(const float* __restrict__ xin,
                                  const float* __restrict__ dlt,
                                  const float* __restrict__ g,
                                  const float* __restrict__ beta,
                                  float* __restrict__ outp,
                                  float* __restrict__ out2) {
  __shared__ float red[256];
  __shared__ float s_mean, s_rstd;
  int row = blockIdx.x;
  int tid = threadIdx.x;
  const float* xr = xin + (long)row * DM;
  const float* dr = dlt + (long)row * DM;
  float v0 = xr[tid] + dr[tid];
  float v1 = xr[tid + 256] + dr[tid + 256];
  red[tid] = v0 + v1;
  __syncthreads();
  for (int s = 128; s > 0; s >>= 1) {
    if (tid < s) red[tid] += red[tid + s];
    __syncthreads();
  }
  if (tid == 0) s_mean = red[0] * (1.f / 512.f);
  __syncthreads();
  float m = s_mean;
  float d0 = v0 - m, d1 = v1 - m;
  red[tid] = d0 * d0 + d1 * d1;
  __syncthreads();
  for (int s = 128; s > 0; s >>= 1) {
    if (tid < s) red[tid] += red[tid + s];
    __syncthreads();
  }
  if (tid == 0) s_rstd = rsqrtf(red[0] * (1.f / 512.f) + 1e-5f);
  __syncthreads();
  float rs = s_rstd;
  float o0 = d0 * rs * g[tid] + beta[tid];
  float o1 = d1 * rs * g[tid + 256] + beta[tid + 256];
  long base = (long)row * DM;
  outp[base + tid] = o0;
  outp[base + tid + 256] = o1;
  if (out2) {
    out2[base + tid] = o0;
    out2[base + tid + 256] = o1;
  }
}

// pred = x @ dec_w + dec_b (N_LEADS=12, VALU is fine)
__global__ void txl_dec_kernel(const float* __restrict__ x,
                               const float* __restrict__ W,
                               const float* __restrict__ bb,
                               float* __restrict__ pred) {
  int row = blockIdx.x;
  int lane = threadIdx.x;
  if (lane < NLEADS) {
    float acc = bb[lane];
    const float* xr = x + (long)row * DM;
    for (int c = 0; c < DM; ++c) acc += xr[c] * W[c * NLEADS + lane];
    pred[(long)row * NLEADS + lane] = acc;
  }
}

// ---------------------------------------------------------------------------
extern "C" void kernel_launch(void* const* d_in, const int* in_sizes, int n_in,
                              void* d_out, int out_size, void* d_ws,
                              size_t ws_size, hipStream_t stream) {
  (void)in_sizes; (void)n_in; (void)out_size; (void)ws_size;
  const float* data  = (const float*)d_in[0];
  const float* mems  = (const float*)d_in[2];
  const float* Wemb  = (const float*)d_in[3];
  const float* bemb  = (const float*)d_in[4];
  const float* qkv_w = (const float*)d_in[5];
  const float* r_w   = (const float*)d_in[6];
  const float* o_w   = (const float*)d_in[7];
  const float* ln1_g = (const float*)d_in[8];
  const float* ln1_b = (const float*)d_in[9];
  const float* ff1_w = (const float*)d_in[10];
  const float* ff1_b = (const float*)d_in[11];
  const float* ff2_w = (const float*)d_in[12];
  const float* ff2_b = (const float*)d_in[13];
  const float* ln2_g = (const float*)d_in[14];
  const float* ln2_b = (const float*)d_in[15];
  const float* dec_w = (const float*)d_in[16];
  const float* dec_b = (const float*)d_in[17];
  const float* rwb   = (const float*)d_in[18];
  const float* rrb   = (const float*)d_in[19];

  float* pred = (float*)d_out;
  float* hids = pred + (long)SEG * BSZ * NLEADS;  // 5 slots of SEG*BSZ*DM

  // workspace carve (f32 then f16), ~97 MB total
  float* ws = (float*)d_ws;
  size_t o = 0;
  float* x   = ws + o; o += (size_t)SEG * BSZ * DM;
  float* pos = ws + o; o += (size_t)KLEN * DM;
  float* vec = ws + o; o += (size_t)SEG * BSZ * DM;
  float* tmp = ws + o; o += (size_t)SEG * BSZ * DM;
  float* x1  = ws + o; o += (size_t)SEG * BSZ * DM;
  float* h   = ws + o; o += (size_t)SEG * BSZ * DI;
  half_t* Qa = (half_t*)(ws + o);
  half_t* Qb = Qa + (size_t)SEG * BSZ * DM;
  half_t* Kf = Qb + (size_t)SEG * BSZ * DM;
  half_t* Vf = Kf + (size_t)KLEN * BSZ * DM;
  half_t* Rk = Vf + (size_t)KLEN * BSZ * DM;

  txl_embed_kernel<<<SEG * BSZ, 256, 0, stream>>>(data, Wemb, bemb, x, hids);
  txl_pos_kernel<<<(KLEN * DM) / 256, 256, 0, stream>>>(pos);

  for (int i = 0; i < NLAYER; ++i) {
    txl_qkv_gemm_kernel<<<dim3(1536 / 64, (KLEN * BSZ) / 16), 32, 0, stream>>>(
        mems + (size_t)i * MEMN * BSZ * DM, x,
        qkv_w + (size_t)i * DM * 1536, rwb, rrb, Qa, Qb, Kf, Vf);
    txl_rk_gemm_kernel<<<dim3(DM / 64, KLEN / 16), 32, 0, stream>>>(
        pos, r_w + (size_t)i * DM * DM, Rk);
    txl_attn_kernel<<<dim3(SEG / 16, BSZ * NH), 32, 0, stream>>>(
        Qa, Qb, Kf, Vf, Rk, vec);
    txl_lin_gemm_kernel<<<dim3(DM / 64, (SEG * BSZ) / 16), 32, 0, stream>>>(
        vec, o_w + (size_t)i * DM * DM, nullptr, tmp, SEG * BSZ, DM, DM, 0);
    txl_add_ln_kernel<<<SEG * BSZ, 256, 0, stream>>>(
        x, tmp, ln1_g + (size_t)i * DM, ln1_b + (size_t)i * DM, x1, nullptr);
    txl_lin_gemm_kernel<<<dim3(DI / 64, (SEG * BSZ) / 16), 32, 0, stream>>>(
        x1, ff1_w + (size_t)i * DM * DI, ff1_b + (size_t)i * DI, h,
        SEG * BSZ, DI, DM, 1);
    txl_lin_gemm_kernel<<<dim3(DM / 64, (SEG * BSZ) / 16), 32, 0, stream>>>(
        h, ff2_w + (size_t)i * DI * DM, ff2_b + (size_t)i * DM, tmp,
        SEG * BSZ, DM, DI, 0);
    txl_add_ln_kernel<<<SEG * BSZ, 256, 0, stream>>>(
        x1, tmp, ln2_g + (size_t)i * DM, ln2_b + (size_t)i * DM, x,
        hids + (size_t)(i + 1) * SEG * BSZ * DM);
  }

  txl_dec_kernel<<<SEG * BSZ, 32, 0, stream>>>(x, dec_w, dec_b, pred);
}
